// BatchTopKSAE_9156870275255
// MI455X (gfx1250) — matrix operand order repaired and checked by reading, compiled, not run on hardware
//
#include <hip/hip_runtime.h>
#include <hip/hip_bf16.h>

#define ACT   768
#define DICT  24576
#define BATCH 4096
#define KBLKS (ACT / 32)          // 24
#define TOPK_TOTAL (32 * 4096)
#define NBINS 8192
#define CAP   2048
#define L1C   0.0008f

typedef float  v8f   __attribute__((ext_vector_type(8)));
typedef __bf16 v16bf __attribute__((ext_vector_type(16)));

union BF16Frag { v16bf v; unsigned short u[16]; uint4 q[2]; };

__device__ __forceinline__ unsigned short f2bf(float f) {
    unsigned int u = __float_as_uint(f);
    unsigned int r = u + 0x7FFFu + ((u >> 16) & 1u);   // round-to-nearest-even
    return (unsigned short)(r >> 16);
}

// ---------------------------------------------------------------- normalize
__global__ void k_normalize(const float* __restrict__ x,
                            const float* __restrict__ b_dec,
                            float* __restrict__ xn,
                            unsigned short* __restrict__ xnbf,
                            float* __restrict__ meanv, float* __restrict__ stdv,
                            float* __restrict__ rs1, float* __restrict__ rs2) {
    int row = blockIdx.x;
    int t = threadIdx.x;
    const float* xr = x + (size_t)row * ACT;
    __shared__ float red[256];

    float v[3];
    float s = 0.f;
    #pragma unroll
    for (int i = 0; i < 3; ++i) { v[i] = xr[t * 3 + i]; s += v[i]; }
    red[t] = s; __syncthreads();
    for (int o = 128; o > 0; o >>= 1) { if (t < o) red[t] += red[t + o]; __syncthreads(); }
    float m = red[0] / (float)ACT;
    __syncthreads();

    float sq = 0.f;
    #pragma unroll
    for (int i = 0; i < 3; ++i) { float d = v[i] - m; sq += d * d; }
    red[t] = sq; __syncthreads();
    for (int o = 128; o > 0; o >>= 1) { if (t < o) red[t] += red[t + o]; __syncthreads(); }
    float sd = sqrtf(red[0] / (float)(ACT - 1));     // unbiased std
    __syncthreads();
    float inv = 1.f / (sd + 1e-5f);

    float s1 = 0.f, s2 = 0.f;
    #pragma unroll
    for (int i = 0; i < 3; ++i) {
        int c = t * 3 + i;
        float xnv = (v[i] - m) * inv;
        xn[(size_t)row * ACT + c] = xnv;
        xnbf[(size_t)row * ACT + c] = f2bf(xnv - b_dec[c]);   // encoder input (xn - b_dec)
        s1 += xnv; s2 += xnv * xnv;
    }
    red[t] = s1; __syncthreads();
    for (int o = 128; o > 0; o >>= 1) { if (t < o) red[t] += red[t + o]; __syncthreads(); }
    if (t == 0) rs1[row] = red[0];
    __syncthreads();
    red[t] = s2; __syncthreads();
    for (int o = 128; o > 0; o >>= 1) { if (t < o) red[t] += red[t + o]; __syncthreads(); }
    if (t == 0) { rs2[row] = red[0]; meanv[row] = m; stdv[row] = sd; }
}

// ---------------------------------------------------------------- W_enc -> bf16, WMMA-B fragment order
// Output layout: wt[(((n_tile * KBLKS) + kblk) * 32 + lane) * 16 + i]
//   = bf16( W_enc[(kblk*32 + (i>>3)*16 + (lane>>4)*8 + (i&7)) * DICT + n_tile*16 + (lane&15)] )
// i.e. each lane's 16 B-fragment elements are contiguous (32 bytes).
__global__ void k_convw(const float* __restrict__ W, unsigned short* __restrict__ wt) {
    __shared__ unsigned short sT[32][72];   // 32(K) x 64(N) bf16 tile, padded
    int nb = blockIdx.x * 64;               // covers 4 n_tiles
    int kb = blockIdx.y * 32;               // one k-block
    int tid = threadIdx.x;

    for (int e = tid; e < 32 * 64; e += 256) {
        int r = e >> 6, c = e & 63;
        sT[r][c] = f2bf(W[(size_t)(kb + r) * DICT + nb + c]);
    }
    __syncthreads();

    int t    = tid >> 6;                    // local n_tile 0..3
    int lane = (tid >> 1) & 31;
    int half = tid & 1;
    int kloc = half * 16 + (lane >> 4) * 8; // K = half*16 + hi*8 + j
    int nloc = t * 16 + (lane & 15);
    unsigned short tmp[8];
    #pragma unroll
    for (int j = 0; j < 8; ++j) tmp[j] = sT[kloc + j][nloc];
    size_t gtile = (size_t)(nb >> 4) + t;
    size_t base = ((gtile * KBLKS + blockIdx.y) * 32 + lane) * 16 + half * 8;
    *(uint4*)(wt + base) = *(const uint4*)tmp;
}

// ---------------------------------------------------------------- encode GEMM (WMMA bf16)
// block = 256 threads = 8 waves. Block tile: 16(M) x 512(N); each wave 4 N-tiles.
// A staged once in LDS (2x ds_load_b128 per frag); B loaded straight from the
// pre-swizzled global buffer (2x global_load_b128 per frag, coalesced). No
// per-iteration barriers.
__global__ __launch_bounds__(256) void
k_encode(const unsigned short* __restrict__ xnbf,
         const unsigned short* __restrict__ wt,
         float* __restrict__ acts) {
    __shared__ unsigned short sA[16 * ACT];   // 24 KB

    const int mBase = blockIdx.y * 16;
    const int tid   = threadIdx.x;
    const int lane  = tid & 31;
    const int wave  = tid >> 5;
    const int hi    = lane >> 4;
    const int lp    = lane & 15;

    {   // stage A: contiguous 16*768 bf16 = 1536 uint4
        const uint4* srcA = (const uint4*)(xnbf + (size_t)mBase * ACT);
        uint4* dA = (uint4*)sA;
        for (int c = tid; c < (16 * ACT) / 8; c += 256) dA[c] = srcA[c];
    }
    __syncthreads();

    const int tile0 = blockIdx.x * 32 + wave * 4;          // first of 4 n_tiles
    const size_t TILE_STRIDE = (size_t)KBLKS * 32 * 16;    // ushorts per n_tile
    const unsigned short* aRow = sA + lp * ACT + hi * 8;
    const unsigned short* bBase =
        wt + ((size_t)tile0 * KBLKS * 32 + lane) * 16;

    v8f acc0 = {}, acc1 = {}, acc2 = {}, acc3 = {};

    #pragma unroll 2
    for (int kblk = 0; kblk < KBLKS; ++kblk) {
        BF16Frag a, b0, b1, b2, b3;
        // A frag: two contiguous 16B runs (K = hi*8+0..7 and 16+hi*8+0..7)
        a.q[0] = *(const uint4*)(aRow + kblk * 32);
        a.q[1] = *(const uint4*)(aRow + kblk * 32 + 16);
        // B frags: contiguous 32B per lane, coalesced across the wave
        const unsigned short* bp = bBase + (size_t)kblk * 512;
        b0.q[0] = ((const uint4*)bp)[0];
        b0.q[1] = ((const uint4*)bp)[1];
        bp += TILE_STRIDE;
        b1.q[0] = ((const uint4*)bp)[0];
        b1.q[1] = ((const uint4*)bp)[1];
        bp += TILE_STRIDE;
        b2.q[0] = ((const uint4*)bp)[0];
        b2.q[1] = ((const uint4*)bp)[1];
        bp += TILE_STRIDE;
        b3.q[0] = ((const uint4*)bp)[0];
        b3.q[1] = ((const uint4*)bp)[1];

        acc0 = __builtin_amdgcn_wmma_f32_16x16x32_bf16(false, a.v, false, b0.v, (short)0, acc0, false, false);
        acc1 = __builtin_amdgcn_wmma_f32_16x16x32_bf16(false, a.v, false, b1.v, (short)0, acc1, false, false);
        acc2 = __builtin_amdgcn_wmma_f32_16x16x32_bf16(false, a.v, false, b2.v, (short)0, acc2, false, false);
        acc3 = __builtin_amdgcn_wmma_f32_16x16x32_bf16(false, a.v, false, b3.v, (short)0, acc3, false, false);
    }

    // epilogue: C/D layout -> (M = r + 8*hi, N = lane&15), fused ReLU
    #pragma unroll
    for (int j = 0; j < 4; ++j) {
        v8f acc = j == 0 ? acc0 : (j == 1 ? acc1 : (j == 2 ? acc2 : acc3));
        int N = (tile0 + j) * 16 + lp;
        #pragma unroll
        for (int r = 0; r < 8; ++r) {
            int M = mBase + r + hi * 8;
            float v = acc[r];
            acts[(size_t)M * DICT + N] = v > 0.f ? v : 0.f;
        }
    }
}

// ---------------------------------------------------------------- histogram + l1/l0
__global__ void k_hist(const float* __restrict__ acts,
                       unsigned int* __restrict__ hist,
                       float* __restrict__ l1sum, unsigned int* __restrict__ l0cnt) {
    __shared__ unsigned int lh[NBINS];        // 32 KB
    __shared__ float redf[256];
    __shared__ unsigned int redu[256];
    int t = threadIdx.x;
    for (int i = t; i < NBINS; i += 256) lh[i] = 0;
    __syncthreads();

    size_t n = (size_t)BATCH * DICT;
    float l1 = 0.f; unsigned int l0 = 0;
    for (size_t i = (size_t)blockIdx.x * 256 + t; i < n; i += (size_t)gridDim.x * 256) {
        float v = acts[i];
        if (v > 0.f) {
            atomicAdd(&lh[__float_as_uint(v) >> 18], 1u);  // monotone key for positives
            l1 += v; ++l0;
        }
    }
    __syncthreads();
    for (int i = t; i < NBINS; i += 256) {
        unsigned int c = lh[i];
        if (c) atomicAdd(&hist[i], c);
    }
    redf[t] = l1; redu[t] = l0; __syncthreads();
    for (int o = 128; o > 0; o >>= 1) {
        if (t < o) { redf[t] += redf[t + o]; redu[t] += redu[t + o]; }
        __syncthreads();
    }
    if (t == 0) { atomicAdd(l1sum, redf[0]); atomicAdd(l0cnt, redu[0]); }
}

// ---------------------------------------------------------------- threshold selection
__global__ void k_threshold(const unsigned int* __restrict__ hist,
                            unsigned int* __restrict__ thresh_bin,
                            unsigned int* __restrict__ budget,
                            unsigned int* __restrict__ claim) {
    __shared__ unsigned int cs[256];
    int t = threadIdx.x;
    unsigned int s = 0;
    for (int i = 0; i < NBINS / 256; ++i) s += hist[t * (NBINS / 256) + i];
    cs[t] = s; __syncthreads();
    if (t == 0) {
        const int CH = NBINS / 256;
        unsigned long long cum = 0; int chunk = -1;
        for (int c = 255; c >= 0; --c) {
            if (cum + cs[c] >= (unsigned long long)TOPK_TOTAL) { chunk = c; break; }
            cum += cs[c];
        }
        unsigned int tb = 0; unsigned long long above = cum;
        if (chunk >= 0) {
            tb = chunk * CH;
            for (int b = chunk * CH + CH - 1; b >= chunk * CH; --b) {
                unsigned int h = hist[b];
                if (above + h >= (unsigned long long)TOPK_TOTAL) { tb = b; break; }
                above += h;
            }
        }
        *thresh_bin = tb;
        unsigned long long bud =
            (unsigned long long)TOPK_TOTAL > above ? (unsigned long long)TOPK_TOTAL - above : 0ull;
        *budget = (unsigned int)bud;
        *claim = 0u;
    }
}

// ---------------------------------------------------------------- top-k scatter + sparse decode
__global__ __launch_bounds__(256) void
k_decode(const float* __restrict__ acts, const float* __restrict__ xn,
         const float* __restrict__ W_dec, const float* __restrict__ b_dec,
         const float* __restrict__ meanv, const float* __restrict__ stdv,
         const unsigned int* __restrict__ thresh_bin,
         const unsigned int* __restrict__ budget,
         unsigned int* __restrict__ claim,
         float* __restrict__ sae_out, float* __restrict__ acts_topk,
         float* __restrict__ row_sqerr) {
    __shared__ unsigned int cols[CAP];
    __shared__ float vals[CAP];
    __shared__ unsigned int cnt;
    __shared__ float red[256];

    int row = blockIdx.x, t = threadIdx.x;
    if (t == 0) cnt = 0;
    __syncthreads();

    unsigned int tb = *thresh_bin, bud = *budget;
    const float* arow = acts + (size_t)row * DICT;
    float* orow = acts_topk + (size_t)row * DICT;

    for (int c = t; c < DICT; c += 256) {
        float v = arow[c];
        bool keep = false;
        if (v > 0.f) {
            unsigned int bin = __float_as_uint(v) >> 18;
            if (bin > tb) keep = true;
            else if (bin == tb) {
                unsigned int p = atomicAdd(claim, 1u);
                keep = (p < bud);
            }
        }
        orow[c] = keep ? v : 0.f;
        if (keep) {
            unsigned int idx = atomicAdd(&cnt, 1u);
            if (idx < CAP) { cols[idx] = (unsigned int)c; vals[idx] = v; }
        }
    }
    __syncthreads();
    unsigned int n = cnt; if (n > CAP) n = CAP;

    int j0 = t * 3;                                 // 768 = 256*3 -> coalesced W_dec reads
    float a0 = b_dec[j0], a1 = b_dec[j0 + 1], a2 = b_dec[j0 + 2];
    for (unsigned int e = 0; e < n; ++e) {
        float v = vals[e];
        const float* wr = W_dec + (size_t)cols[e] * ACT;
        a0 = fmaf(v, wr[j0],     a0);
        a1 = fmaf(v, wr[j0 + 1], a1);
        a2 = fmaf(v, wr[j0 + 2], a2);
    }

    float m = meanv[row], sd = stdv[row];
    const float* xnr = xn + (size_t)row * ACT;
    float e0 = a0 - xnr[j0], e1 = a1 - xnr[j0 + 1], e2 = a2 - xnr[j0 + 2];
    float sq = e0 * e0 + e1 * e1 + e2 * e2;

    float* so = sae_out + (size_t)row * ACT;
    so[j0]     = a0 * sd + m;
    so[j0 + 1] = a1 * sd + m;
    so[j0 + 2] = a2 * sd + m;

    red[t] = sq; __syncthreads();
    for (int o = 128; o > 0; o >>= 1) { if (t < o) red[t] += red[t + o]; __syncthreads(); }
    if (t == 0) row_sqerr[row] = red[0];
}

// ---------------------------------------------------------------- scalar losses
__global__ void k_final(const float* __restrict__ row_sqerr,
                        const float* __restrict__ rs1, const float* __restrict__ rs2,
                        const float* __restrict__ l1sum, const unsigned int* __restrict__ l0cnt,
                        float* __restrict__ out_scalars) {
    __shared__ float r0[256], r1[256], r2[256];
    int t = threadIdx.x;
    float sse = 0.f, s1 = 0.f, s2 = 0.f;
    for (int i = t; i < BATCH; i += 256) { sse += row_sqerr[i]; s1 += rs1[i]; s2 += rs2[i]; }
    r0[t] = sse; r1[t] = s1; r2[t] = s2; __syncthreads();
    for (int o = 128; o > 0; o >>= 1) {
        if (t < o) { r0[t] += r0[t + o]; r1[t] += r1[t + o]; r2[t] += r2[t + o]; }
        __syncthreads();
    }
    if (t == 0) {
        double N = (double)BATCH * ACT;
        float l2   = (float)((double)r0[0] / N);
        float xvar = (float)(((double)r2[0] - (double)r1[0] * (double)r1[0] / N) / (N - 1.0));
        float fvu  = l2 / (xvar + 1e-10f);
        float l1n  = *l1sum / (float)BATCH;
        float l0n  = (float)(*l0cnt) / (float)BATCH;
        float l1l  = L1C * l1n;
        out_scalars[0] = l2 + l1l;  // loss
        out_scalars[1] = l2;        // l2_loss
        out_scalars[2] = l1l;       // l1_loss
        out_scalars[3] = l0n;       // l0_norm
        out_scalars[4] = l1n;       // l1_norm
        out_scalars[5] = fvu;       // fvu
    }
}

// ---------------------------------------------------------------- launch
extern "C" void kernel_launch(void* const* d_in, const int* in_sizes, int n_in,
                              void* d_out, int out_size, void* d_ws, size_t ws_size,
                              hipStream_t stream) {
    (void)in_sizes; (void)n_in; (void)out_size; (void)ws_size;
    const float* x     = (const float*)d_in[0];
    const float* W_enc = (const float*)d_in[1];
    const float* W_dec = (const float*)d_in[2];
    const float* b_dec = (const float*)d_in[3];
    // d_in[4] = top_k (== 32, baked into TOPK_TOTAL)

    char* ws = (char*)d_ws;
    size_t off = 0;
    auto alloc = [&](size_t bytes) -> char* {
        char* p = ws + off;
        off += (bytes + 255) & ~(size_t)255;
        return p;
    };
    float*          xn    = (float*)alloc((size_t)BATCH * ACT * 4);
    unsigned short* xnbf  = (unsigned short*)alloc((size_t)BATCH * ACT * 2);
    unsigned short* wt    = (unsigned short*)alloc((size_t)ACT * DICT * 2);
    float*          acts  = (float*)alloc((size_t)BATCH * DICT * 4);
    float*          meanv = (float*)alloc(BATCH * 4);
    float*          stdv  = (float*)alloc(BATCH * 4);
    float*          rs1   = (float*)alloc(BATCH * 4);
    float*          rs2   = (float*)alloc(BATCH * 4);
    float*          rsq   = (float*)alloc(BATCH * 4);
    char* hregion = alloc(NBINS * 4 + 64);
    unsigned int* hist  = (unsigned int*)hregion;
    float*        l1sum = (float*)(hregion + NBINS * 4);
    unsigned int* l0cnt = (unsigned int*)(hregion + NBINS * 4 + 4);
    unsigned int* tb    = (unsigned int*)(hregion + NBINS * 4 + 8);
    unsigned int* bud   = (unsigned int*)(hregion + NBINS * 4 + 12);
    unsigned int* claim = (unsigned int*)(hregion + NBINS * 4 + 16);

    hipMemsetAsync(hregion, 0, NBINS * 4 + 64, stream);

    float* sae_out   = (float*)d_out;
    float* acts_topk = sae_out + (size_t)BATCH * ACT;
    float* scal      = acts_topk + (size_t)BATCH * DICT;

    k_normalize<<<BATCH, 256, 0, stream>>>(x, b_dec, xn, xnbf, meanv, stdv, rs1, rs2);

    dim3 gw(DICT / 64, ACT / 32);
    k_convw<<<gw, 256, 0, stream>>>(W_enc, wt);

    dim3 g(DICT / 512, BATCH / 16);
    k_encode<<<g, 256, 0, stream>>>(xnbf, wt, acts);

    k_hist<<<2048, 256, 0, stream>>>(acts, hist, l1sum, l0cnt);
    k_threshold<<<1, 256, 0, stream>>>(hist, tb, bud, claim);
    k_decode<<<BATCH, 256, 0, stream>>>(acts, xn, W_dec, b_dec, meanv, stdv,
                                        tb, bud, claim, sae_out, acts_topk, rsq);
    k_final<<<1, 256, 0, stream>>>(rsq, rs1, rs2, l1sum, l0cnt, scal);
}
